// RockInstanceNetSparse_46737834115411
// MI455X (gfx1250) — matrix operand order repaired and checked by reading, not compiled, not run on hardware
//
#include <hip/hip_runtime.h>
#include <hip/hip_bf16.h>

// CDNA5 / gfx1250: wave32, WMMA 16x16x32 bf16 -> f32 accumulate.
typedef __attribute__((ext_vector_type(16))) __bf16 v16bf;
typedef __attribute__((ext_vector_type(8)))  float  v8f;

#define C_DIM 64
#define HEADS 8
#define HDIM  8
#define KNN   16

// ---------------------------------------------------------------------------
// Packed weight layout (bf16): [m(4)][ktile(2)][ntile(4)][p:hi/lo(2)][lane(32)][e(16)]
// For V_WMMA B-matrix (32x16, 16-bit): lane L holds column N = L&15,
// element e corresponds to K = (e>>3)*16 + (L>>4)*8 + (e&7).  One contiguous 32B read per lane.
// ---------------------------------------------------------------------------
__device__ __host__ __forceinline__ int pw_index(int m, int j, int t, int p, int lane) {
  return ((((m * 2 + j) * 4 + t) * 2 + p) * 32 + lane) * 16;
}

__global__ __launch_bounds__(256) void prep_weights_kernel(
    const float* __restrict__ Wq, const float* __restrict__ Wk,
    const float* __restrict__ Wv, const float* __restrict__ Wo,
    __bf16* __restrict__ packed)
{
  int id = blockIdx.x * 256 + threadIdx.x;          // 4*2*4*32*16 = 16384
  if (id >= 4 * 2 * 4 * 32 * 16) return;
  int e    = id & 15;
  int lane = (id >> 4) & 31;
  int t    = (id >> 9) & 3;
  int j    = (id >> 11) & 1;
  int m    = id >> 12;
  const float* W = (m == 0) ? Wq : (m == 1) ? Wk : (m == 2) ? Wv : Wo;
  int kk = j * 32 + (e >> 3) * 16 + (lane >> 4) * 8 + (e & 7); // K index within 64
  int nn = t * 16 + (lane & 15);                    // N index within 64
  float w = W[kk * C_DIM + nn];
  __bf16 hi = (__bf16)w;
  __bf16 lo = (__bf16)(w - (float)hi);              // error-compensation term
  packed[pw_index(m, j, t, 0, lane) + e] = hi;
  packed[pw_index(m, j, t, 1, lane) + e] = lo;
}

// ---------------------------------------------------------------------------
// A-fragment loader: V_WMMA A-matrix (16x32, 16-bit) layout:
// lane L holds row M = L&15; elements 0..7  -> K = j*32 + (L>>4)*8 + (0..7)
//                            elements 8..15 -> K = j*32 + 16 + (L>>4)*8 + (0..7)
// Source is f32; split into bf16 hi + lo.
// ---------------------------------------------------------------------------
__device__ __forceinline__ void load_a_frag(const float* __restrict__ rowp,
                                            int j, int kgrp,
                                            v16bf& ahi, v16bf& alo)
{
  const float* p0 = rowp + j * 32 + kgrp * 8;
  float fa[16];
  float4 f;
  f = *(const float4*)(p0 + 0);  fa[0]=f.x;  fa[1]=f.y;  fa[2]=f.z;  fa[3]=f.w;
  f = *(const float4*)(p0 + 4);  fa[4]=f.x;  fa[5]=f.y;  fa[6]=f.z;  fa[7]=f.w;
  f = *(const float4*)(p0 + 16); fa[8]=f.x;  fa[9]=f.y;  fa[10]=f.z; fa[11]=f.w;
  f = *(const float4*)(p0 + 20); fa[12]=f.x; fa[13]=f.y; fa[14]=f.z; fa[15]=f.w;
#pragma unroll
  for (int e = 0; e < 16; ++e) {
    __bf16 h = (__bf16)fa[e];
    ahi[e] = h;
    alo[e] = (__bf16)(fa[e] - (float)h);
  }
}

__device__ __forceinline__ v8f wmma_bf16(v16bf a, v16bf b, v8f c) {
  // 8 args: (neg_a, A, neg_b, B, c_mod, C, reuse_a, reuse_b)
  return __builtin_amdgcn_wmma_f32_16x16x32_bf16(false, a, false, b,
                                                 (short)0, c, false, false);
}

// ---------------------------------------------------------------------------
// Fused QKV projection: one wave computes a 16-row x 64-col block for q,k,v.
// feats[16x64] (fp32 -> bf16 hi/lo) x W[64x64] with f32 WMMA accumulation.
// ---------------------------------------------------------------------------
__global__ __launch_bounds__(256) void qkv_gemm_kernel(
    const float* __restrict__ feats, const __bf16* __restrict__ packed,
    float* __restrict__ q, float* __restrict__ k, float* __restrict__ v, int N)
{
  int wave    = blockIdx.x * (256 / 32) + (threadIdx.x >> 5);
  int nBlocks = (N + 15) >> 4;
  if (wave >= nBlocks) return;
  int lane    = threadIdx.x & 31;
  int kgrp    = lane >> 4;
  int rowBase = wave << 4;
  int row     = rowBase + (lane & 15);
  if (row >= N) row = N - 1;                        // N=100000 is 16-aligned anyway
  const float* rowp = feats + (size_t)row * C_DIM;

  v16bf ahi[2], alo[2];
  load_a_frag(rowp, 0, kgrp, ahi[0], alo[0]);
  load_a_frag(rowp, 1, kgrp, ahi[1], alo[1]);

  float* outs[3] = {q, k, v};
#pragma unroll
  for (int m = 0; m < 3; ++m) {
#pragma unroll
    for (int t = 0; t < 4; ++t) {
      v8f acc = {};
#pragma unroll
      for (int j = 0; j < 2; ++j) {
        v16bf bhi = *(const v16bf*)(packed + pw_index(m, j, t, 0, lane));
        v16bf blo = *(const v16bf*)(packed + pw_index(m, j, t, 1, lane));
        acc = wmma_bf16(ahi[j], bhi, acc);   // hi*hi
        acc = wmma_bf16(ahi[j], blo, acc);   // hi*lo
        acc = wmma_bf16(alo[j], bhi, acc);   // lo*hi  (~fp32-quality product)
      }
      // D layout: VGPR r, lanes 0-15: M=r, N=lane; lanes 16-31: M=8+r.
      int col = t * 16 + (lane & 15);
      int r0  = rowBase + kgrp * 8;
      float* op = outs[m] + (size_t)r0 * C_DIM + col;
#pragma unroll
      for (int r = 0; r < 8; ++r)
        if (r0 + r < N) op[(size_t)r * C_DIM] = acc[r];
    }
  }
}

// ---------------------------------------------------------------------------
// Neighborhood attention: one thread per (point, head).
// Gathers 16 neighbors' k/v rows (L2-resident), stable softmax over K.
// ---------------------------------------------------------------------------
__global__ __launch_bounds__(256) void attn_kernel(
    const float* __restrict__ q, const float* __restrict__ kf,
    const float* __restrict__ vf, const int* __restrict__ knn,
    float* __restrict__ attn_out, int total)
{
  int tid = blockIdx.x * 256 + threadIdx.x;
  if (tid >= total) return;
  int n = tid >> 3;
  int h = tid & 7;

  const float* qp = q + (size_t)n * C_DIM + h * HDIM;
  float4 q0 = *(const float4*)qp;
  float4 q1 = *(const float4*)(qp + 4);

  const int* ip = knn + (size_t)n * KNN;
  float sc[KNN];
  float mx = -3.0e38f;
#pragma unroll
  for (int kk = 0; kk < KNN; ++kk) {
    int j = ip[kk];
    const float* kp = kf + (size_t)j * C_DIM + h * HDIM;
    __builtin_prefetch(vf + (size_t)j * C_DIM + h * HDIM, 0, 0); // warm v row
    float4 k0 = *(const float4*)kp;
    float4 k1 = *(const float4*)(kp + 4);
    float s = q0.x*k0.x + q0.y*k0.y + q0.z*k0.z + q0.w*k0.w
            + q1.x*k1.x + q1.y*k1.y + q1.z*k1.z + q1.w*k1.w;
    s *= 0.35355339059327373f;                     // 1/sqrt(8)
    sc[kk] = s;
    mx = fmaxf(mx, s);
  }
  float sum = 0.0f;
#pragma unroll
  for (int kk = 0; kk < KNN; ++kk) {
    float e = __expf(sc[kk] - mx);
    sc[kk] = e;
    sum += e;
  }
  float inv = 1.0f / sum;

  float a0=0,a1=0,a2=0,a3=0,a4=0,a5=0,a6=0,a7=0;
#pragma unroll
  for (int kk = 0; kk < KNN; ++kk) {
    int j = ip[kk];
    const float* vp = vf + (size_t)j * C_DIM + h * HDIM;
    float4 v0 = *(const float4*)vp;
    float4 v1 = *(const float4*)(vp + 4);
    float w = sc[kk] * inv;
    a0 += w * v0.x; a1 += w * v0.y; a2 += w * v0.z; a3 += w * v0.w;
    a4 += w * v1.x; a5 += w * v1.y; a6 += w * v1.z; a7 += w * v1.w;
  }
  float* op = attn_out + (size_t)n * C_DIM + h * HDIM;
  float4 o0 = {a0, a1, a2, a3};
  float4 o1 = {a4, a5, a6, a7};
  *(float4*)op       = o0;
  *(float4*)(op + 4) = o1;
}

// ---------------------------------------------------------------------------
// Output projection: attn[N,64] x Wo[64,64] + bo -> out (same WMMA scheme).
// ---------------------------------------------------------------------------
__global__ __launch_bounds__(256) void oproj_gemm_kernel(
    const float* __restrict__ attn, const __bf16* __restrict__ packed,
    const float* __restrict__ bo, float* __restrict__ out, int N)
{
  int wave    = blockIdx.x * (256 / 32) + (threadIdx.x >> 5);
  int nBlocks = (N + 15) >> 4;
  if (wave >= nBlocks) return;
  int lane    = threadIdx.x & 31;
  int kgrp    = lane >> 4;
  int rowBase = wave << 4;
  int row     = rowBase + (lane & 15);
  if (row >= N) row = N - 1;
  const float* rowp = attn + (size_t)row * C_DIM;

  v16bf ahi[2], alo[2];
  load_a_frag(rowp, 0, kgrp, ahi[0], alo[0]);
  load_a_frag(rowp, 1, kgrp, ahi[1], alo[1]);

#pragma unroll
  for (int t = 0; t < 4; ++t) {
    v8f acc = {};
#pragma unroll
    for (int j = 0; j < 2; ++j) {
      v16bf bhi = *(const v16bf*)(packed + pw_index(3, j, t, 0, lane));
      v16bf blo = *(const v16bf*)(packed + pw_index(3, j, t, 1, lane));
      acc = wmma_bf16(ahi[j], bhi, acc);
      acc = wmma_bf16(ahi[j], blo, acc);
      acc = wmma_bf16(alo[j], bhi, acc);
    }
    int col   = t * 16 + (lane & 15);
    float bia = bo[col];
    int r0    = rowBase + kgrp * 8;
    float* op = out + (size_t)r0 * C_DIM + col;
#pragma unroll
    for (int r = 0; r < 8; ++r)
      if (r0 + r < N) op[(size_t)r * C_DIM] = acc[r] + bia;
  }
}

// ---------------------------------------------------------------------------
extern "C" void kernel_launch(void* const* d_in, const int* in_sizes, int n_in,
                              void* d_out, int out_size, void* d_ws, size_t ws_size,
                              hipStream_t stream)
{
  const float* feats = (const float*)d_in[0];
  const int*   knn   = (const int*)  d_in[1];
  const float* Wq    = (const float*)d_in[2];
  const float* Wk    = (const float*)d_in[3];
  const float* Wv    = (const float*)d_in[4];
  const float* Wo    = (const float*)d_in[5];
  const float* bo    = (const float*)d_in[6];
  float*       out   = (float*)d_out;

  const int N = in_sizes[0] / C_DIM;               // 100000

  // Workspace layout: [packed weights 64KB][q][k][v][attn] (all f32, N*64 each)
  char*   ws     = (char*)d_ws;
  __bf16* packed = (__bf16*)ws;
  float*  q      = (float*)(ws + 65536);
  float*  k      = q + (size_t)N * C_DIM;
  float*  v      = k + (size_t)N * C_DIM;
  float*  attn   = v + (size_t)N * C_DIM;

  // 1) repack weights into WMMA B-fragment layout (bf16 hi/lo)
  prep_weights_kernel<<<64, 256, 0, stream>>>(Wq, Wk, Wv, Wo, packed);

  // 2) fused QKV projection (WMMA)
  int nBlocks = (N + 15) / 16;                     // 16-row blocks, 1 wave each
  int grid1   = (nBlocks + 7) / 8;                 // 8 waves per 256-thread block
  qkv_gemm_kernel<<<grid1, 256, 0, stream>>>(feats, packed, q, k, v, N);

  // 3) k-NN attention
  int total = N * HEADS;
  attn_kernel<<<(total + 255) / 256, 256, 0, stream>>>(q, k, v, knn, attn, total);

  // 4) output projection + bias (WMMA)
  oproj_gemm_kernel<<<grid1, 256, 0, stream>>>(attn, packed, bo, out, N);
}
